// DpEmbeddingNet_36790689857887
// MI455X (gfx1250) — matrix operand order repaired and checked by reading, not compiled, not run on hardware
//
#include <hip/hip_runtime.h>

typedef __attribute__((ext_vector_type(16))) _Float16 v16h;
typedef __attribute__((ext_vector_type(8)))  float    v8f;

#define N_ELEM  (32 * 192 * 128)   // 786432 pointwise elements
#define NTILES  (N_ELEM / 16)      // 49152 M-tiles of 16 elements
#define NWAVES  8                  // 256 threads / wave32
#define NBLOCKS 768                // 768*8 waves -> 8 grid-stride iters each

// 16-bit A-matrix 16x32 fragment K-index for vector element h (ISA 7.12.2):
// lanes 0-15 hold K=0..7,16..23 ; lanes 16-31 hold K=8..15,24..31
__device__ __forceinline__ int kmapA(int h, int laneHi) {
    return (h & 7) + (laneHi ? 8 : 0) + ((h < 8) ? 0 : 16);
}

// Branch-free tanh: tanh(x) = 1 - 2/(exp(2x)+1).
// Lowers to v_exp_f32 + v_add + v_rcp_f32 + v_fma (TRANS32 ops co-execute
// with WMMA). Saturates correctly to +/-1; accurate to ~1e-7 absolute for
// the small pre-activations this net produces (far below the f16 input
// quantization of the WMMA path).
__device__ __forceinline__ float fast_tanh(float x) {
    float e = __builtin_amdgcn_exp2f(x * 2.885390081777927f); // exp(2x)
    float r = __builtin_amdgcn_rcpf(e + 1.0f);
    return __builtin_fmaf(r, -2.0f, 1.0f);
}

__global__ __launch_bounds__(256)
void dp_mlp_wmma_kernel(const float* __restrict__ tilde_r,
                        const float* __restrict__ w0, const float* __restrict__ b0,
                        const float* __restrict__ w1, const float* __restrict__ b1,
                        const float* __restrict__ w2, const float* __restrict__ b2,
                        float* __restrict__ out)
{
    // Zero-padded f16 weights resident in LDS (WGP has 320KB; we use ~35KB)
    __shared__ _Float16 sw1[32][64];      // w1: 25x50 -> 32x64
    __shared__ _Float16 sw2[64][112];     // w2: 50x100 -> 64x112
    __shared__ float    sb1[64];
    __shared__ float    sb2[112];
    __shared__ float    sw0[32];
    __shared__ float    sb0[32];
    // Wave-private transpose staging for layer-1 activations (row pad 64->72
    // f16 breaks the 128B-row bank alias on the A-fragment gather)
    __shared__ _Float16 stage[NWAVES][16][72];

    const int tid = threadIdx.x;

    for (int i = tid; i < 32 * 64; i += 256) {
        int k = i >> 6, n = i & 63;
        sw1[k][n] = (_Float16)((k < 25 && n < 50) ? w1[k * 50 + n] : 0.0f);
    }
    for (int i = tid; i < 64 * 112; i += 256) {
        int k = i / 112, n = i % 112;
        sw2[k][n] = (_Float16)((k < 50 && n < 100) ? w2[k * 100 + n] : 0.0f);
    }
    for (int i = tid; i < 64; i += 256)  sb1[i] = (i < 50)  ? b1[i] : 0.0f;
    for (int i = tid; i < 112; i += 256) sb2[i] = (i < 100) ? b2[i] : 0.0f;
    for (int i = tid; i < 32; i += 256) {
        sw0[i] = (i < 25) ? w0[i] : 0.0f;
        sb0[i] = (i < 25) ? b0[i] : 0.0f;
    }
    __syncthreads();

    const int lane   = tid & 31;
    const int wid    = tid >> 5;
    const int lmod   = lane & 15;
    const int laneHi = lane >> 4;

    // ---- Persistent B fragments (built once, reused for all M-tiles) ----
    // B (KxN=32x16) f16 layout: lane holds column n=lane%16; element h holds
    // K = kmapA(h, laneHi) (same K order as the A fragment).
    v16h b1f[4];
#pragma unroll
    for (int j = 0; j < 4; ++j) {
#pragma unroll
        for (int h = 0; h < 16; ++h) {
            int k = kmapA(h, laneHi);
            b1f[j][h] = sw1[k][j * 16 + lmod];
        }
    }
    v16h b2f[7][2];
#pragma unroll
    for (int j = 0; j < 7; ++j) {
#pragma unroll
        for (int kf = 0; kf < 2; ++kf) {
#pragma unroll
            for (int h = 0; h < 16; ++h) {
                int k = kf * 32 + kmapA(h, laneHi);
                b2f[j][kf][h] = sw2[k][j * 16 + lmod];
            }
        }
    }

    for (int tile = blockIdx.x * NWAVES + wid; tile < NTILES;
         tile += gridDim.x * NWAVES) {

        // ---- Layer 0: x -> h0 (built directly in A-fragment layout) ----
        // Each lane owns row m = lane%16 of the tile; both half-waves load
        // the same 16 rows (required by the interleaved A layout).
        float x = __builtin_nontemporal_load(
                      &tilde_r[(size_t)(tile * 16 + lmod) * 4]); // channel 0
        v16h a1;
#pragma unroll
        for (int h = 0; h < 16; ++h) {
            int k = kmapA(h, laneHi);          // k>=25 -> tanh(0)=0 (padded)
            a1[h] = (_Float16)fast_tanh(__builtin_fmaf(x, sw0[k], sb0[k]));
        }

        // ---- Layer 1: 16x32 @ 32x64 via 4 WMMAs, tanh, stage to LDS ----
#pragma unroll
        for (int j = 0; j < 4; ++j) {
            v8f c = {};
            c = __builtin_amdgcn_wmma_f32_16x16x32_f16(
                    false, a1, false, b1f[j], (short)0, c, false, false);
            int n = j * 16 + lmod;                // D layout: lane = column
#pragma unroll
            for (int r = 0; r < 8; ++r) {
                int m = r + laneHi * 8;           // D layout: VGPR r = row
                stage[wid][m][n] = (_Float16)fast_tanh(c[r] + sb1[n]);
            }
        }

        // ---- Transpose bounce: gather layer-1 output as A fragments ----
        v16h a2[2];
#pragma unroll
        for (int kf = 0; kf < 2; ++kf) {
#pragma unroll
            for (int h = 0; h < 16; ++h) {
                int k = kf * 32 + kmapA(h, laneHi);
                a2[kf][h] = stage[wid][lmod][k];
            }
        }

        // ---- Layer 2: 16x64 @ 64x112 via 7 x (2 chained WMMAs) ----
#pragma unroll
        for (int j = 0; j < 7; ++j) {
            v8f c = {};
            c = __builtin_amdgcn_wmma_f32_16x16x32_f16(
                    false, a2[0], false, b2f[j][0], (short)0, c, false, false);
            c = __builtin_amdgcn_wmma_f32_16x16x32_f16(
                    false, a2[1], false, b2f[j][1], (short)0, c, false, false);
            int n = j * 16 + lmod;
            if (n < 100) {                        // predicate only the stores
#pragma unroll
                for (int r = 0; r < 8; ++r) {
                    int m = r + laneHi * 8;
                    // streaming store: 315MB written once, never re-read
                    __builtin_nontemporal_store(
                        fast_tanh(c[r] + sb2[n]),
                        &out[(size_t)(tile * 16 + m) * 100 + n]);
                }
            }
        }
    }
}

extern "C" void kernel_launch(void* const* d_in, const int* in_sizes, int n_in,
                              void* d_out, int out_size, void* d_ws, size_t ws_size,
                              hipStream_t stream) {
    (void)in_sizes; (void)n_in; (void)d_ws; (void)ws_size; (void)out_size;
    const float* tilde_r = (const float*)d_in[0];
    const float* w0 = (const float*)d_in[1];
    const float* b0 = (const float*)d_in[2];
    const float* w1 = (const float*)d_in[3];
    const float* b1 = (const float*)d_in[4];
    const float* w2 = (const float*)d_in[5];
    const float* b2 = (const float*)d_in[6];
    float* out = (float*)d_out;

    dp_mlp_wmma_kernel<<<NBLOCKS, 256, 0, stream>>>(
        tilde_r, w0, b0, w1, b1, w2, b2, out);
}
